// Philter_1812476199726
// MI455X (gfx1250) — matrix operand order repaired and checked by reading, not compiled, not run on hardware
//
#include <hip/hip_runtime.h>
#include <hip/hip_bf16.h>
#include <math.h>

// ---------------------------------------------------------------------------
// Reference:  Sinkhorn(log|W|+eps, 50 iters) -> Wc ;  out = (x @ Wc^T) * PHI
// Log-domain Sinkhorn iterates are la0 + u_i + v_j  =>  only two 1024-vector
// potentials are updated; the 4MB matrix is read (L2-resident) per pass.
// GEMM: v_wmma_f32_16x16x32_bf16 with split-bf16 (hi+lo, 3 passes) for
// ~fp32 accuracy at bf16-WMMA throughput, register-staged software pipeline
// so global loads of chunk k+1 overlap the 24 WMMAs of chunk k.
// ---------------------------------------------------------------------------

typedef __attribute__((ext_vector_type(16))) __bf16 v16bf;
typedef __attribute__((ext_vector_type(8)))  float  v8f;

#define PHI_F 1.6180339887498949f
#define DIM   1024
#define MROWS 16384
#define NITER 50

// ---------------- prep: la0 = log(|W|+eps), plus transposed copy ------------
__global__ __launch_bounds__(256) void prep_kernel(const float* __restrict__ W,
                                                   float* __restrict__ LA,
                                                   float* __restrict__ LAT,
                                                   float* __restrict__ u,
                                                   float* __restrict__ v) {
  __shared__ float tile[32][33];
  const int tx = threadIdx.x, ty = threadIdx.y;
  const int x  = blockIdx.x * 32 + tx;
  const int y0 = blockIdx.y * 32 + ty;
#pragma unroll
  for (int j = 0; j < 32; j += 8) {
    float w  = W[(size_t)(y0 + j) * DIM + x];
    float la = logf(fabsf(w) + 1e-8f);
    LA[(size_t)(y0 + j) * DIM + x] = la;
    tile[ty + j][tx] = la;
  }
  __syncthreads();
  const int xt  = blockIdx.y * 32 + tx;
  const int yt0 = blockIdx.x * 32 + ty;
#pragma unroll
  for (int j = 0; j < 32; j += 8)
    LAT[(size_t)(yt0 + j) * DIM + xt] = tile[tx][ty + j];

  if (blockIdx.x == 0 && blockIdx.y == 0) {
    const int t = ty * 32 + tx;
    for (int i = t; i < DIM; i += 256) { u[i] = 0.f; v[i] = 0.f; }
  }
}

// ---- one Sinkhorn half-step: out[r] = -logsumexp_j(M[r,j] + add[j]) --------
__global__ __launch_bounds__(256) void lse_pass_kernel(const float* __restrict__ Msrc,
                                                       const float* __restrict__ addv,
                                                       float* __restrict__ outv) {
  __shared__ float red[256];
  const int row = blockIdx.x;
  const int t   = threadIdx.x;
  const float* pr = Msrc + (size_t)row * DIM;

  float vals[4];
  float mx = -3.4e38f;
#pragma unroll
  for (int k = 0; k < 4; ++k) {
    int j = t + k * 256;
    vals[k] = pr[j] + addv[j];
    mx = fmaxf(mx, vals[k]);
  }
  red[t] = mx;
  __syncthreads();
#pragma unroll
  for (int s = 128; s > 0; s >>= 1) {
    if (t < s) red[t] = fmaxf(red[t], red[t + s]);
    __syncthreads();
  }
  const float rmax = red[0];
  __syncthreads();

  float sum = 0.f;
#pragma unroll
  for (int k = 0; k < 4; ++k) sum += expf(vals[k] - rmax);
  red[t] = sum;
  __syncthreads();
#pragma unroll
  for (int s = 128; s > 0; s >>= 1) {
    if (t < s) red[t] += red[t + s];
    __syncthreads();
  }
  if (t == 0) outv[row] = -(rmax + logf(red[0]));
}

// -------- split fp32 -> bf16 hi/lo pair (classic two-term split) ------------
__device__ __forceinline__ void split_bf16(float f, __bf16& hi, __bf16& lo) {
  hi = (__bf16)f;
  lo = (__bf16)(f - (float)hi);
}

__global__ __launch_bounds__(256) void split_x_kernel(const float* __restrict__ x,
                                                      __bf16* __restrict__ xhi,
                                                      __bf16* __restrict__ xlo) {
  const size_t idx = (size_t)blockIdx.x * 256 + threadIdx.x;
  split_bf16(x[idx], xhi[idx], xlo[idx]);
}

// WT[j,i] = exp(LAT[j,i] + u[i] + v[j])  == Wc^T row-major, split bf16
__global__ __launch_bounds__(256) void build_wt_kernel(const float* __restrict__ LAT,
                                                       const float* __restrict__ u,
                                                       const float* __restrict__ v,
                                                       __bf16* __restrict__ wthi,
                                                       __bf16* __restrict__ wtlo) {
  const size_t idx = (size_t)blockIdx.x * 256 + threadIdx.x;
  const int j = (int)(idx >> 10);   // row of LAT  (k index of GEMM)
  const int i = (int)(idx & 1023);  // col of LAT  (n index of GEMM)
  const float val = expf(LAT[idx] + u[i] + v[j]);
  split_bf16(val, wthi[idx], wtlo[idx]);
}

// ---------------------------- WMMA GEMM -------------------------------------
// C[m,n] = PHI * sum_k A[m,k] * B[k,n],  A = x (M x K), B = Wc^T (K x N)
// 128x128 tile / block, 8 waves (4 m-groups x 2 n-groups), K chunked by 32.
// Register-staged pipeline: global loads for chunk k+1 are in flight during
// the WMMAs of chunk k (s_wait_loadcnt lands at next iteration's ds_store).
#define TM 128
#define TN 128
#define TK 32
#define LDA_P 34    // TK + 2 bf16 pad (68B row stride -> conflict-free b32 reads)
#define LDB_P 130   // TN + 2 bf16 pad

__device__ __forceinline__ void st_lds8(__bf16* dst, uint4 w) {
  unsigned int* d = (unsigned int*)dst;  // 4B-aligned (even bf16 index)
  d[0] = w.x; d[1] = w.y; d[2] = w.z; d[3] = w.w;
}

__global__ __launch_bounds__(256) void gemm_bf16x3_kernel(
    const __bf16* __restrict__ Ahi, const __bf16* __restrict__ Alo,
    const __bf16* __restrict__ Bhi, const __bf16* __restrict__ Blo,
    float* __restrict__ C, int M, int N, int K) {
  __shared__ __bf16 sAh[TM * LDA_P];
  __shared__ __bf16 sAl[TM * LDA_P];
  __shared__ __bf16 sBh[TK * LDB_P];
  __shared__ __bf16 sBl[TK * LDB_P];

  const int tid  = threadIdx.x;
  const int lane = tid & 31;
  const int wid  = tid >> 5;
  const int wm   = wid & 3;   // wave's 32-row group within the 128-row tile
  const int wn   = wid >> 2;  // wave's 64-col group within the 128-col tile
  const int tileM = blockIdx.y * TM;
  const int tileN = blockIdx.x * TN;

  const v8f vzero = {};
  v8f acc[2][4];
#pragma unroll
  for (int mt = 0; mt < 2; ++mt)
#pragma unroll
    for (int nt = 0; nt < 4; ++nt) acc[mt][nt] = vzero;

  // cooperative tile loaders: 16 contiguous bf16 (2x b128) per thread/matrix
  const int arow  = tid >> 1;          // 0..127
  const int ahalf = (tid & 1) * 16;    // 0 or 16
  const int brow  = tid >> 3;          // 0..31
  const int bseg  = (tid & 7) * 16;    // 0..112

  const int lg8 = ((lane >> 4) & 1) << 3;  // 0 for lanes 0-15, 8 for 16-31
  const int ml  = lane & 15;

  uint4 rAh[2], rAl[2], rBh[2], rBl[2];  // staged chunk (in-flight loads)

  auto load_stage = [&](int kc) {
    const uint4* pAh = (const uint4*)(Ahi + (size_t)(tileM + arow) * K + kc + ahalf);
    const uint4* pAl = (const uint4*)(Alo + (size_t)(tileM + arow) * K + kc + ahalf);
    const uint4* pBh = (const uint4*)(Bhi + (size_t)(kc + brow) * N + tileN + bseg);
    const uint4* pBl = (const uint4*)(Blo + (size_t)(kc + brow) * N + tileN + bseg);
    rAh[0] = pAh[0]; rAh[1] = pAh[1];
    rAl[0] = pAl[0]; rAl[1] = pAl[1];
    rBh[0] = pBh[0]; rBh[1] = pBh[1];
    rBl[0] = pBl[0]; rBl[1] = pBl[1];
  };

  load_stage(0);  // prologue

  for (int kk = 0; kk < K; kk += TK) {
    // drain staged registers into LDS (waits on the in-flight loads here)
    st_lds8(&sAh[arow * LDA_P + ahalf],     rAh[0]);
    st_lds8(&sAh[arow * LDA_P + ahalf + 8], rAh[1]);
    st_lds8(&sAl[arow * LDA_P + ahalf],     rAl[0]);
    st_lds8(&sAl[arow * LDA_P + ahalf + 8], rAl[1]);
    st_lds8(&sBh[brow * LDB_P + bseg],      rBh[0]);
    st_lds8(&sBh[brow * LDB_P + bseg + 8],  rBh[1]);
    st_lds8(&sBl[brow * LDB_P + bseg],      rBl[0]);
    st_lds8(&sBl[brow * LDB_P + bseg + 8],  rBl[1]);

    // kick off next chunk's global loads; they stay in flight across compute
    if (kk + TK < K) {
      load_stage(kk + TK);
      if (kk + 2 * TK < K) {  // warm L2 two chunks ahead (global_prefetch_b8)
        __builtin_prefetch(Ahi + (size_t)(tileM + arow) * K + kk + 2 * TK + ahalf, 0, 3);
        __builtin_prefetch(Bhi + (size_t)(kk + 2 * TK + brow) * N + tileN + bseg, 0, 3);
      }
    }
    __syncthreads();

    // A fragments (ISA 7.12.2, 16-bit A 16x32): lane -> M, VGPR/pack -> K
    v16bf fAh[2], fAl[2];
#pragma unroll
    for (int mt = 0; mt < 2; ++mt) {
      const int mrow = wm * 32 + mt * 16 + ml;
      const __bf16* pAh = &sAh[mrow * LDA_P];
      const __bf16* pAl = &sAl[mrow * LDA_P];
#pragma unroll
      for (int vv = 0; vv < 8; ++vv) {
        const int k = ((vv & 4) << 2) + lg8 + ((vv & 3) << 1);
        fAh[mt][2 * vv]     = pAh[k];
        fAh[mt][2 * vv + 1] = pAh[k + 1];
        fAl[mt][2 * vv]     = pAl[k];
        fAl[mt][2 * vv + 1] = pAl[k + 1];
      }
    }

#pragma unroll
    for (int nt = 0; nt < 4; ++nt) {
      // B fragment (16-bit B 32x16): lane -> N, VGPR/pack -> K
      v16bf fBh, fBl;
      const __bf16* pBh = &sBh[wn * 64 + nt * 16 + ml];
      const __bf16* pBl = &sBl[wn * 64 + nt * 16 + ml];
#pragma unroll
      for (int vv = 0; vv < 8; ++vv) {
        const int k = ((vv & 4) << 2) + lg8 + ((vv & 3) << 1);
        fBh[2 * vv]     = pBh[k * LDB_P];
        fBh[2 * vv + 1] = pBh[(k + 1) * LDB_P];
        fBl[2 * vv]     = pBl[k * LDB_P];
        fBl[2 * vv + 1] = pBl[(k + 1) * LDB_P];
      }
#pragma unroll
      for (int mt = 0; mt < 2; ++mt) {
        acc[mt][nt] = __builtin_amdgcn_wmma_f32_16x16x32_bf16(
            false, fAh[mt], false, fBh, (short)0, acc[mt][nt], false, false);
        acc[mt][nt] = __builtin_amdgcn_wmma_f32_16x16x32_bf16(
            false, fAh[mt], false, fBl, (short)0, acc[mt][nt], false, false);
        acc[mt][nt] = __builtin_amdgcn_wmma_f32_16x16x32_bf16(
            false, fAl[mt], false, fBh, (short)0, acc[mt][nt], false, false);
      }
    }
    __syncthreads();  // LDS consumed; safe to overwrite next iteration
  }

  // Epilogue: f32 C/D layout — VGPR r: M=r (lanes 0-15), M=8+r (lanes 16-31)
  const int nl = lane & 15;
  const int mh = ((lane >> 4) & 1) << 3;
#pragma unroll
  for (int mt = 0; mt < 2; ++mt) {
#pragma unroll
    for (int nt = 0; nt < 4; ++nt) {
      const int row0 = tileM + wm * 32 + mt * 16 + mh;
      const int col  = tileN + wn * 64 + nt * 16 + nl;
#pragma unroll
      for (int r = 0; r < 8; ++r)
        C[(size_t)(row0 + r) * N + col] = PHI_F * acc[mt][nt][r];
    }
  }
}

// ---------------------------------------------------------------------------
extern "C" void kernel_launch(void* const* d_in, const int* in_sizes, int n_in,
                              void* d_out, int out_size, void* d_ws, size_t ws_size,
                              hipStream_t stream) {
  const float* x = (const float*)d_in[0];  // [16384, 1024]
  const float* W = (const float*)d_in[1];  // [1024, 1024]
  float* out = (float*)d_out;              // [16384, 1024]

  char* ws = (char*)d_ws;
  float* LA   = (float*)(ws);                              //  4 MiB
  float* LAT  = (float*)(ws + (size_t)(4 << 20));          //  4 MiB
  float* u    = (float*)(ws + (size_t)(8 << 20));          //  4 KiB
  float* v    = u + DIM;                                   //  4 KiB
  __bf16* wthi = (__bf16*)(ws + (size_t)(8 << 20) + (1 << 16));  // 2 MiB
  __bf16* wtlo = wthi + (size_t)DIM * DIM;                        // 2 MiB
  __bf16* xhi  = (__bf16*)(ws + (size_t)(16 << 20));              // 32 MiB
  __bf16* xlo  = xhi + (size_t)MROWS * DIM;                       // 32 MiB

  // 1) la0, la0^T, zero potentials
  prep_kernel<<<dim3(DIM / 32, DIM / 32), dim3(32, 8), 0, stream>>>(W, LA, LAT, u, v);

  // 2) split x into bf16 hi/lo
  split_x_kernel<<<(MROWS * DIM) / 256, 256, 0, stream>>>(x, xhi, xlo);

  // 3) 50 Sinkhorn iterations on the potentials only (matrix stays in L2)
  for (int it = 0; it < NITER; ++it) {
    lse_pass_kernel<<<DIM, 256, 0, stream>>>(LA,  v, u);  // u = -LSE_j(la0 + v)
    lse_pass_kernel<<<DIM, 256, 0, stream>>>(LAT, u, v);  // v = -LSE_i(la0 + u)
  }

  // 4) materialize Wc^T = exp(la0^T + u + v) as split bf16
  build_wt_kernel<<<(DIM * DIM) / 256, 256, 0, stream>>>(LAT, u, v, wthi, wtlo);

  // 5) out = PHI * x @ Wc^T via bf16x3 WMMA
  gemm_bf16x3_kernel<<<dim3(DIM / TN, MROWS / TM), 256, 0, stream>>>(
      xhi, xlo, wthi, wtlo, out, MROWS, DIM, DIM);
}
